// Siglip2Attention_2216203125185
// MI455X (gfx1250) — matrix operand order repaired and checked by reading, not compiled, not run on hardware
//
#include <hip/hip_runtime.h>
#include <hip/hip_bf16.h>
#include <stdint.h>

typedef __bf16 bf16_t;
typedef __attribute__((ext_vector_type(16))) __bf16 v16bf;
typedef __attribute__((ext_vector_type(8)))  float  v8f;

#ifndef __has_builtin
#define __has_builtin(x) 0
#endif

#if __has_builtin(__builtin_amdgcn_global_load_async_to_lds_b128)
#define USE_ASYNC_LDS 1
#else
#define USE_ASYNC_LDS 0
#endif

constexpr int SEQ  = 4096;
constexpr int EMB  = 1152;
constexpr int NH   = 16;
constexpr int HD   = 72;
constexpr int NSEG = 8;
constexpr int DQK  = 96;   // head dim padded to 3 x 32 for QK^T k-loop
constexpr int DV   = 80;   // head dim padded to 5 x 16 for PV n-tiles
constexpr int LSTR = 40;   // LDS row stride (elements): 80B, 16B-aligned, bank-conflict-free
constexpr float SM_SCALE = 0.11785113019775793f; // 1/sqrt(72)

// ---------------- WMMA helpers (CDNA5 wave32, 16x16x32 bf16 -> f32) --------

static __device__ inline v8f wmma_bf16(v16bf a, v16bf b, v8f c) {
  return __builtin_amdgcn_wmma_f32_16x16x32_bf16(
      /*neg_a=*/false, a, /*neg_b=*/false, b,
      /*c_mod=*/(short)0, c, /*reuse_a=*/false, /*reuse_b=*/false);
}

// A fragment 16x32: rowp points at (m=lane&15, k0); reg i -> k=(i&3)*2+(i>>2)*16+8g
static __device__ inline v16bf load_frag_a(const bf16_t* rowp, int g) {
  union { v16bf v; unsigned u[8]; } r;
  const unsigned* p = (const unsigned*)rowp;
#pragma unroll
  for (int i = 0; i < 8; ++i) r.u[i] = p[(i & 3) + ((i >> 2) * 8) + 4 * g];
  return r.v;
}

// B fragment 32x16: rowp points at (n=lane&15, k0) of a row-major [n][k] buffer;
// reg i -> k=(i&3)*2+(i>>2)*16+8g (same as A fragment)
static __device__ inline v16bf load_frag_b(const bf16_t* rowp, int g) {
  union { v16bf v; unsigned u[8]; } r;
  const unsigned* p = (const unsigned*)rowp;
#pragma unroll
  for (int i = 0; i < 8; ++i) r.u[i] = p[(i & 3) + ((i >> 2) * 8) + 4 * g];
  return r.v;
}

#if USE_ASYNC_LDS
// Builtin prototype (from diagnostic): (v4i AS1*, v4i AS3*, imm offset, imm cpol)
typedef int v4i_t __attribute__((__vector_size__(16)));
typedef __attribute__((address_space(1))) v4i_t* glob_v4i_p;
typedef __attribute__((address_space(3))) v4i_t* lds_v4i_p;
static __device__ inline glob_v4i_p to_glob4(const void* p) {
  return (glob_v4i_p)(uintptr_t)p;
}
static __device__ inline lds_v4i_p to_lds4(void* p) {
  return (lds_v4i_p)(uint32_t)(uintptr_t)p;   // low 32 bits of generic ptr = LDS offset
}
#define ASYNC_COPY16(gsrc, ldst) \
  __builtin_amdgcn_global_load_async_to_lds_b128(to_glob4(gsrc), to_lds4(ldst), 0, 0)
#define ASYNC_WAIT() asm volatile("s_wait_asynccnt 0x0" ::: "memory")
#else
#define ASYNC_WAIT() do {} while (0)
#endif

// ---------------- elementwise prep kernels ---------------------------------

__global__ void cvt_f32_bf16(const float* __restrict__ in, bf16_t* __restrict__ out, int n) {
  for (int i = blockIdx.x * blockDim.x + threadIdx.x; i < n; i += gridDim.x * blockDim.x)
    out[i] = (bf16_t)in[i];
}

__global__ void seg_build(const int* __restrict__ cu, int* __restrict__ seg) {
  int i = blockIdx.x * blockDim.x + threadIdx.x;
  if (i < SEQ) {
    int s = 0;
#pragma unroll
    for (int j = 0; j < NSEG + 1; ++j) s += (cu[j] <= i) ? 1 : 0;  // searchsorted right
    seg[i] = s;
  }
}

// RoPE on q/k, scatter q/k to [H][SEQ][96] (zero pad 72..95), v to d-major [H][80][SEQ]
__global__ void rope_scatter(const bf16_t* __restrict__ qkv,
                             const float* __restrict__ cosT, const float* __restrict__ sinT,
                             bf16_t* __restrict__ Qb, bf16_t* __restrict__ Kb,
                             bf16_t* __restrict__ Vt) {
  int idx = blockIdx.x * blockDim.x + threadIdx.x;   // SEQ*NH*48 threads
  int t   = idx % 48;
  int h   = (idx / 48) % NH;
  int pos = idx / (48 * NH);
  if (pos >= SEQ) return;
  size_t qoff = ((size_t)h * SEQ + pos) * DQK;
  if (t < 36) {
    const bf16_t* row = qkv + (size_t)pos * (3 * EMB);
    int   base = h * HD + t;
    float c = cosT[pos * HD + t], s = sinT[pos * HD + t];
    float qa = (float)row[base],            qbv = (float)row[base + 36];
    float ka = (float)row[EMB + base],      kbv = (float)row[EMB + base + 36];
    float va = (float)row[2 * EMB + base],  vbv = (float)row[2 * EMB + base + 36];
    float qra = qa * c - qbv * s, qrb = qbv * c + qa * s;
    float kra = ka * c - kbv * s, krb = kbv * c + ka * s;
    Qb[qoff + t]      = (bf16_t)(qra * SM_SCALE);
    Qb[qoff + t + 36] = (bf16_t)(qrb * SM_SCALE);
    Kb[qoff + t]      = (bf16_t)kra;
    Kb[qoff + t + 36] = (bf16_t)krb;
    Vt[((size_t)h * DV + t)      * SEQ + pos] = (bf16_t)va;
    Vt[((size_t)h * DV + t + 36) * SEQ + pos] = (bf16_t)vbv;
  } else {
    int u = t - 36;                 // 0..11 -> zero pad
    int d = HD + 2 * u;             // 72..94
    Qb[qoff + d] = (bf16_t)0.f; Qb[qoff + d + 1] = (bf16_t)0.f;
    Kb[qoff + d] = (bf16_t)0.f; Kb[qoff + d + 1] = (bf16_t)0.f;
    if (u < 4) {
      Vt[((size_t)h * DV + HD + 2 * u)     * SEQ + pos] = (bf16_t)0.f;
      Vt[((size_t)h * DV + HD + 2 * u + 1) * SEQ + pos] = (bf16_t)0.f;
    }
  }
}

// ---------------- GEMM: C[m][n] = sum_k A[m][k]*W[n][k] + bias[n] ----------
// Block tile 256x128 (8 waves as 4Mx2N, wave tile 64x64 = 16 WMMA accums).
// K-step 32 staged in LDS (A 256x32, B 128x32), double-buffered; async-to-LDS
// copies (ASYNCcnt) when available, global->reg->ds_store fallback otherwise.

static __device__ inline void stage_tile(const bf16_t* __restrict__ Ag,
                                         const bf16_t* __restrict__ Wg,
                                         int K, int k0,
                                         bf16_t (*lsA)[LSTR], bf16_t (*lsB)[LSTR],
                                         int tid) {
  const bf16_t* arow = Ag + (size_t)tid * K + k0;                       // 256 rows of A
  const bf16_t* brow = Wg + (size_t)(tid & 127) * K + k0 + (tid >> 7) * 16;  // 128 rows of W
  bf16_t* la = &lsA[tid][0];
  bf16_t* lb = &lsB[tid & 127][(tid >> 7) * 16];
#if USE_ASYNC_LDS
#pragma unroll
  for (int i = 0; i < 4; ++i)
    ASYNC_COPY16(arow + 8 * i, la + 8 * i);
  ASYNC_COPY16(brow, lb);
  ASYNC_COPY16(brow + 8, lb + 8);
#else
  uint4 a0 = *(const uint4*)(arow);
  uint4 a1 = *(const uint4*)(arow + 8);
  uint4 a2 = *(const uint4*)(arow + 16);
  uint4 a3 = *(const uint4*)(arow + 24);
  uint4 b0 = *(const uint4*)(brow);
  uint4 b1 = *(const uint4*)(brow + 8);
  *(uint4*)(la)      = a0;
  *(uint4*)(la + 8)  = a1;
  *(uint4*)(la + 16) = a2;
  *(uint4*)(la + 24) = a3;
  *(uint4*)(lb)      = b0;
  *(uint4*)(lb + 8)  = b1;
#endif
}

__global__ __launch_bounds__(256) void gemm_bf16_lds(
    const bf16_t* __restrict__ A, const bf16_t* __restrict__ W,
    const float* __restrict__ bias, bf16_t* __restrict__ outb,
    float* __restrict__ outf, int M, int N, int K) {
  __shared__ __align__(16) bf16_t lsA[2][256][LSTR];   // 40 KB
  __shared__ __align__(16) bf16_t lsB[2][128][LSTR];   // 20 KB
  const int tid  = threadIdx.x;
  const int lane = tid & 31;
  const int wid  = tid >> 5;
  const int nn   = lane & 15;
  const int g    = lane >> 4;
  const int wr   = wid >> 1;          // 0..3 -> M offset wr*64
  const int wc   = wid & 1;           // 0..1 -> N offset wc*64
  const int bm   = blockIdx.x * 256;
  const int bn   = blockIdx.y * 128;
  const bf16_t* Ag = A + (size_t)bm * K;
  const bf16_t* Wg = W + (size_t)bn * K;

  v8f acc[4][4];
#pragma unroll
  for (int i = 0; i < 4; ++i)
#pragma unroll
    for (int j = 0; j < 4; ++j) { v8f z = {}; acc[i][j] = z; }

  stage_tile(Ag, Wg, K, 0, lsA[0], lsB[0], tid);
  const int nsteps = K / 32;
  for (int s = 0; s < nsteps; ++s) {
    const int buf = s & 1;
    ASYNC_WAIT();
    __syncthreads();                                   // staged tile visible to all waves
    if (s + 1 < nsteps)
      stage_tile(Ag, Wg, K, (s + 1) * 32, lsA[buf ^ 1], lsB[buf ^ 1], tid);
    v16bf af[4];
#pragma unroll
    for (int i = 0; i < 4; ++i)
      af[i] = load_frag_a(&lsA[buf][wr * 64 + 16 * i + nn][0], g);
#pragma unroll
    for (int j = 0; j < 4; ++j) {
      v16bf bfj = load_frag_b(&lsB[buf][wc * 64 + 16 * j + nn][0], g);
#pragma unroll
      for (int i = 0; i < 4; ++i)
        acc[i][j] = wmma_bf16(af[i], bfj, acc[i][j]);
    }
  }

#pragma unroll
  for (int i = 0; i < 4; ++i)
#pragma unroll
    for (int j = 0; j < 4; ++j) {
      int col = bn + wc * 64 + 16 * j + nn;
      float bv = bias[col];
#pragma unroll
      for (int r = 0; r < 8; ++r) {
        int row = bm + wr * 64 + 16 * i + 8 * g + r;
        float v = acc[i][j][r] + bv;
        if (outb) outb[(size_t)row * N + col] = (bf16_t)v;
        else      outf[(size_t)row * N + col] = v;
      }
    }
}

// ---------------- flash attention: 1 wave per 16-row query tile ------------

__global__ __launch_bounds__(256) void attn_flash(
    const bf16_t* __restrict__ Qb, const bf16_t* __restrict__ Kb,
    const bf16_t* __restrict__ Vt, const int* __restrict__ seg,
    const int* __restrict__ cu, bf16_t* __restrict__ attnb) {
  __shared__ __align__(16) bf16_t lds_p[8][16][36];   // per-wave P tile (16 rows x 32 keys)
  const int lane = threadIdx.x & 31;
  const int w    = threadIdx.x >> 5;
  const int nn   = lane & 15;
  const int g    = lane >> 4;
  const int h    = blockIdx.y;
  const int r0   = blockIdx.x * 128 + w * 16;

  const bf16_t* qbase  = Qb + ((size_t)h * SEQ + r0) * DQK;
  const bf16_t* khead  = Kb + (size_t)h * SEQ * DQK;
  const bf16_t* vhead  = Vt + (size_t)h * DV * SEQ;

  v16bf aq[3];
#pragma unroll
  for (int kc = 0; kc < 3; ++kc)
    aq[kc] = load_frag_a(qbase + (size_t)nn * DQK + kc * 32, g);

  int segq[8];
#pragma unroll
  for (int r = 0; r < 8; ++r) segq[r] = seg[r0 + 8 * g + r];

  const int kstart = cu[seg[r0] - 1];
  const int kend   = cu[seg[r0 + 15]];

  float mrun[8], lrun[8];
  v8f acc[5];
#pragma unroll
  for (int r = 0; r < 8; ++r) { mrun[r] = -3.0e38f; lrun[r] = 0.f; }
#pragma unroll
  for (int dt = 0; dt < 5; ++dt) { v8f z = {}; acc[dt] = z; }

  for (int kb = (kstart & ~31); kb < kend; kb += 32) {
    // ---- scores for two 16-key subtiles (K-dim = 96 padded) ----
    v8f sc[2];
    int  segk[2]; bool kval[2];
#pragma unroll
    for (int t = 0; t < 2; ++t) {
      int key  = kb + 16 * t + nn;
      int keyc = min(key, SEQ - 1);
      kval[t]  = (key < kend);
      segk[t]  = seg[keyc];
      v8f z = {}; sc[t] = z;
      const bf16_t* krow = khead + (size_t)keyc * DQK;
#pragma unroll
      for (int kc = 0; kc < 3; ++kc)
        sc[t] = wmma_bf16(aq[kc], load_frag_b(krow + kc * 32, g), sc[t]);
    }
    // ---- online softmax over the 32-key chunk ----
#pragma unroll
    for (int r = 0; r < 8; ++r) {
      bool v0 = kval[0] && (segk[0] == segq[r]);
      bool v1 = kval[1] && (segk[1] == segq[r]);
      float x0 = v0 ? sc[0][r] : -3.0e38f;
      float x1 = v1 ? sc[1][r] : -3.0e38f;
      float xm = fmaxf(x0, x1);
#pragma unroll
      for (int off = 8; off; off >>= 1) xm = fmaxf(xm, __shfl_xor(xm, off, 32));
      float nm    = fmaxf(mrun[r], xm);
      float alpha = __expf(mrun[r] - nm);
      float p0 = v0 ? __expf(x0 - nm) : 0.f;
      float p1 = v1 ? __expf(x1 - nm) : 0.f;
      float ps = p0 + p1;
#pragma unroll
      for (int off = 8; off; off >>= 1) ps += __shfl_xor(ps, off, 32);
      lrun[r] = lrun[r] * alpha + ps;
      mrun[r] = nm;
#pragma unroll
      for (int dt = 0; dt < 5; ++dt) acc[dt][r] *= alpha;
      lds_p[w][r + 8 * g][nn]      = (bf16_t)p0;   // C-layout -> [row][key] stage
      lds_p[w][r + 8 * g][16 + nn] = (bf16_t)p1;
    }
    asm volatile("s_wait_dscnt 0x0" ::: "memory");  // wave-local LDS RAW fence
    // ---- O += P @ V (5 d-tiles of 16) ----
    v16bf ap = load_frag_a(&lds_p[w][nn][0], g);
#pragma unroll
    for (int dt = 0; dt < 5; ++dt)
      acc[dt] = wmma_bf16(ap, load_frag_b(vhead + (size_t)(dt * 16 + nn) * SEQ + kb, g),
                          acc[dt]);
  }

  // ---- normalize + write bf16 attention output [SEQ][EMB] ----
#pragma unroll
  for (int r = 0; r < 8; ++r) {
    float inv = 1.0f / lrun[r];
    int row = r0 + 8 * g + r;
#pragma unroll
    for (int dt = 0; dt < 5; ++dt) {
      int d = dt * 16 + nn;
      if (d < HD)
        attnb[(size_t)row * EMB + h * HD + d] = (bf16_t)(acc[dt][r] * inv);
    }
  }
}

// ---------------- launch ---------------------------------------------------

extern "C" void kernel_launch(void* const* d_in, const int* in_sizes, int n_in,
                              void* d_out, int out_size, void* d_ws, size_t ws_size,
                              hipStream_t stream) {
  const float* hidden = (const float*)d_in[0];
  const int*   cu     = (const int*)d_in[1];
  const float* cosT   = (const float*)d_in[2];
  const float* sinT   = (const float*)d_in[3];
  const float* w_qkv  = (const float*)d_in[4];
  const float* b_qkv  = (const float*)d_in[5];
  const float* w_out  = (const float*)d_in[6];
  const float* b_out  = (const float*)d_in[7];
  float* out = (float*)d_out;
  (void)in_sizes; (void)n_in; (void)out_size; (void)ws_size;

  char* ws = (char*)d_ws;
  size_t off = 0;
  auto alloc = [&](size_t bytes) -> void* {
    void* p = ws + off;
    off += (bytes + 255) & ~(size_t)255;
    return p;
  };
  bf16_t* hb    = (bf16_t*)alloc((size_t)SEQ * EMB * 2);
  bf16_t* wqkvb = (bf16_t*)alloc((size_t)3 * EMB * EMB * 2);
  bf16_t* woutb = (bf16_t*)alloc((size_t)EMB * EMB * 2);
  bf16_t* qkvb  = (bf16_t*)alloc((size_t)SEQ * 3 * EMB * 2);
  bf16_t* Qb    = (bf16_t*)alloc((size_t)NH * SEQ * DQK * 2);
  bf16_t* Kb    = (bf16_t*)alloc((size_t)NH * SEQ * DQK * 2);
  bf16_t* Vt    = (bf16_t*)alloc((size_t)NH * DV * SEQ * 2);
  bf16_t* attnb = (bf16_t*)alloc((size_t)SEQ * EMB * 2);
  int*    segp  = (int*)alloc((size_t)SEQ * 4);

  // 1) fp32 -> bf16 operand conversion
  cvt_f32_bf16<<<4096, 256, 0, stream>>>(hidden, hb, SEQ * EMB);
  cvt_f32_bf16<<<4096, 256, 0, stream>>>(w_qkv, wqkvb, 3 * EMB * EMB);
  cvt_f32_bf16<<<2048, 256, 0, stream>>>(w_out, woutb, EMB * EMB);
  // 2) segment ids
  seg_build<<<(SEQ + 255) / 256, 256, 0, stream>>>(cu, segp);
  // 3) QKV GEMM (+bias) -> bf16 qkv
  gemm_bf16_lds<<<dim3(SEQ / 256, (3 * EMB) / 128), 256, 0, stream>>>(
      hb, wqkvb, b_qkv, qkvb, nullptr, SEQ, 3 * EMB, EMB);
  // 4) RoPE + scatter (scale folded into Q)
  rope_scatter<<<(SEQ * NH * 48) / 256, 256, 0, stream>>>(qkvb, cosT, sinT, Qb, Kb, Vt);
  // 5) block-diagonal flash attention
  attn_flash<<<dim3(SEQ / 128, NH), 256, 0, stream>>>(Qb, Kb, Vt, segp, cu, attnb);
  // 6) output projection (+bias) -> fp32
  gemm_bf16_lds<<<dim3(SEQ / 256, EMB / 128), 256, 0, stream>>>(
      attnb, woutb, b_out, nullptr, out, SEQ, EMB, EMB);
}
